// EqualizedModConv2D_42090679500960
// MI455X (gfx1250) — matrix operand order repaired and checked by reading, not compiled, not run on hardware
//
#include <hip/hip_runtime.h>
#include <hip/hip_bf16.h>

typedef __attribute__((ext_vector_type(16))) __bf16 v16bf;
typedef __attribute__((ext_vector_type(8)))  __bf16 v8bf;
typedef __attribute__((ext_vector_type(8)))  float  v8f;

#define NBATCH 16
#define IC 512
#define OC 512
#define HW 32
#define PIX 1024          // 32*32
#define PH 34             // padded height
#define PW 34             // padded width

// ---------------- mod[n,ic] = style @ (fc_w * fc_scale)^T + bias + 1 ----------------
__global__ void mod_kernel(const float* __restrict__ style,
                           const float* __restrict__ fcw,
                           const float* __restrict__ bias,
                           float* __restrict__ mod) {
  int idx = blockIdx.x * blockDim.x + threadIdx.x;   // 16*512 = 8192
  int n = idx >> 9, ic = idx & 511;
  const float fc_scale = 0.044194173824159216f;      // 1/sqrt(512)
  const float* s = style + n * 512;
  const float* w = fcw + ic * 512;
  float acc = 0.f;
  for (int d = 0; d < 512; ++d) acc += s[d] * w[d];
  mod[idx] = acc * fc_scale + bias[ic] + 1.0f;
}

// ------------- wsq[oc,ic] = sum_k w^2 ;  wT[kpos][oc][ic] = bf16(w) -------------
__global__ void wprep_kernel(const float* __restrict__ weight,
                             float* __restrict__ wsq,
                             __bf16* __restrict__ wT) {
  int idx = blockIdx.x * blockDim.x + threadIdx.x;   // 512*512 = 262144
  int oc = idx >> 9, ic = idx & 511;
  float s = 0.f;
#pragma unroll
  for (int k = 0; k < 9; ++k) {
    float v = weight[idx * 9 + k];
    s += v * v;
    wT[k * (OC * IC) + oc * 512 + ic] = (__bf16)v;
  }
  wsq[idx] = s;
}

// ------- demod[n,oc] = rsqrt( w_scale^2 * sum_ic mod^2 * wsq + eps ) -------
__global__ void demod_kernel(const float* __restrict__ mod,
                             const float* __restrict__ wsq,
                             float* __restrict__ demod) {
  int idx = blockIdx.x * blockDim.x + threadIdx.x;   // 16*512
  int n = idx >> 9, oc = idx & 511;
  const float* m = mod + n * 512;
  const float* q = wsq + oc * 512;
  float s = 0.f;
  for (int ic = 0; ic < 512; ++ic) { float mv = m[ic]; s += mv * mv * q[ic]; }
  const float ws2 = 1.0f / 4608.0f;                  // (1/sqrt(IC*9))^2
  demod[idx] = rsqrtf(ws2 * s + 1e-8f);
}

// ---- xmp[n][hh][ww][ic] = bf16( x[n,ic,hh-1,ww-1] * mod[n,ic] ), zero border ----
__global__ void xpad_kernel(const float* __restrict__ x,
                            const float* __restrict__ mod,
                            __bf16* __restrict__ xmp) {
  int idx = blockIdx.x * blockDim.x + threadIdx.x;   // 16*34*34*512 = 9469952
  int ic = idx & 511;
  int r  = idx >> 9;
  int ww = r % PW;  r /= PW;
  int hh = r % PH;
  int n  = r / PH;
  int h = hh - 1, w = ww - 1;
  float v = 0.f;
  if ((unsigned)h < 32u && (unsigned)w < 32u)
    v = x[((n * IC + ic) * HW + h) * HW + w] * mod[n * IC + ic];
  xmp[idx] = (__bf16)v;
}

// ---------------- LDS-free implicit-GEMM 3x3 conv with bf16 WMMA ----------------
// Block: 256 oc x 64 pixels (2 image rows) of one sample; 256 threads = 8 waves.
// Wave w owns TWO oc strips {16w, 16w+128} x 4 pixel subtiles = 8 C tiles.
// B fragments are register-reused across both strips: 12 loads per 8 WMMAs.
__global__ __launch_bounds__(256) void conv_kernel(
    const __bf16* __restrict__ xmp,    // [N][34][34][IC]  (pre-modulated, padded)
    const __bf16* __restrict__ wT,     // [9][OC][IC]
    const float*  __restrict__ demod,  // [N][OC]
    float*        __restrict__ out) {  // [N][OC][32][32]
  const int tid  = threadIdx.x;
  const int lane = tid & 31, wave = tid >> 5;
  const int g    = lane >> 4, lm = lane & 15;
  const int pixbase = blockIdx.x * 64;          // 16 tiles of 64 pixels
  const int ocbase  = blockIdx.y * 256;         // 2 tiles of 256 oc
  const int n       = blockIdx.z;
  const int h0      = pixbase >> 5;             // first output row of tile

  v8f acc[2][4] = {};

  // A row this lane feeds for strip 0; strip 1 is +128 rows (constant offset)
  const int ocr = ocbase + wave * 16 + lm;
  const int AS  = 128 * IC;                     // strip-1 A offset (elements)

  // nt pixel offsets in elements of xmp (w+16, next row, next row + 16)
  const int NT1 = 16 * IC;          //  8192
  const int NT2 = PW * IC;          // 17408
  const int NT3 = (PW + 16) * IC;   // 25600

  for (int kpos = 0; kpos < 9; ++kpos) {
    const int kh = kpos / 3, kw = kpos % 3;
    // A: elems 0..7 = K{8g..8g+7}, elems 8..15 = K{16+8g..16+8g+7}
    const __bf16* aptr = wT + ((size_t)(kpos * OC + ocr)) * IC + 8 * g;
    // B: elems 0..7 = K{8g..8g+7}, elems 8..15 = K{16+8g..16+8g+7} at fixed pixel (h0+kh, lm+kw)
    const __bf16* bptr =
        xmp + (((size_t)n * PH + (h0 + kh)) * PW + (lm + kw)) * IC + 8 * g;

#pragma unroll 2
    for (int ks = 0; ks < 16; ++ks) {           // 16 K-chunks of 32 channels
      v16bf a[2];
      {
        v8bf a00 = *(const v8bf*)(aptr);
        v8bf a01 = *(const v8bf*)(aptr + 16);
        v8bf a10 = *(const v8bf*)(aptr + AS);
        v8bf a11 = *(const v8bf*)(aptr + AS + 16);
#pragma unroll
        for (int i = 0; i < 8; ++i) {
          a[0][i] = a00[i]; a[0][8 + i] = a01[i];
          a[1][i] = a10[i]; a[1][8 + i] = a11[i];
        }
      }

#pragma unroll
      for (int nt = 0; nt < 4; ++nt) {
        const int off = (nt == 0) ? 0 : (nt == 1) ? NT1 : (nt == 2) ? NT2 : NT3;
        v8bf b0 = *(const v8bf*)(bptr + off);
        v8bf b1 = *(const v8bf*)(bptr + off + 16);
        v16bf b;
#pragma unroll
        for (int i = 0; i < 8; ++i) { b[i] = b0[i]; b[8 + i] = b1[i]; }
        acc[0][nt] = __builtin_amdgcn_wmma_f32_16x16x32_bf16(
            false, a[0], false, b, (short)0, acc[0][nt], false, false);
        acc[1][nt] = __builtin_amdgcn_wmma_f32_16x16x32_bf16(
            false, a[1], false, b, (short)0, acc[1][nt], false, false);
      }
      aptr += 32;
      bptr += 32;
    }
  }

  // ---- epilogue: scale by w_scale * demod[n,oc] and store ----
  const float w_scale = 0.014731391274719739f;  // 1/sqrt(4608)
#pragma unroll
  for (int ms = 0; ms < 2; ++ms) {
#pragma unroll
    for (int r = 0; r < 8; ++r) {
      int oc = ocbase + ms * 128 + wave * 16 + 8 * g + r;  // C layout: M = 8g + r
      float s = w_scale * demod[n * OC + oc];
#pragma unroll
      for (int nt = 0; nt < 4; ++nt) {
        int pix = pixbase + nt * 16 + lm;                  // N = lane & 15
        out[(size_t)(n * OC + oc) * PIX + pix] = acc[ms][nt][r] * s;
      }
    }
  }
}

extern "C" void kernel_launch(void* const* d_in, const int* in_sizes, int n_in,
                              void* d_out, int out_size, void* d_ws, size_t ws_size,
                              hipStream_t stream) {
  const float* x      = (const float*)d_in[0];  // (16,512,32,32)
  const float* style  = (const float*)d_in[1];  // (16,512)
  const float* weight = (const float*)d_in[2];  // (512,512,3,3)
  const float* fcw    = (const float*)d_in[3];  // (512,512)
  const float* bias   = (const float*)d_in[4];  // (512,)
  float* out = (float*)d_out;

  char* ws = (char*)d_ws;
  float*  mod   = (float*)(ws);                                   // 32 KB
  float*  dmod  = (float*)(ws + (32u << 10));                     // 32 KB
  float*  wsq   = (float*)(ws + (64u << 10));                     // 1 MB
  __bf16* wT    = (__bf16*)(ws + (64u << 10) + (1u << 20));       // 4.5 MB
  __bf16* xmp   = (__bf16*)(ws + (64u << 10) + (1u << 20) + 9u * OC * IC * 2u); // ~18 MB

  mod_kernel  <<<32,    256, 0, stream>>>(style, fcw, bias, mod);
  wprep_kernel<<<1024,  256, 0, stream>>>(weight, wsq, wT);
  demod_kernel<<<32,    256, 0, stream>>>(mod, wsq, dmod);
  xpad_kernel <<<36992, 256, 0, stream>>>(x, mod, xmp);           // 16*34*34*512
  conv_kernel <<<dim3(16, 2, 16), 256, 0, stream>>>(xmp, wT, dmod, out);
}
